// DeformableAttention_48515950575963
// MI455X (gfx1250) — matrix operand order repaired and checked by reading, not compiled, not run on hardware
//
#include <hip/hip_runtime.h>
#include <hip/hip_bf16.h>

typedef __attribute__((ext_vector_type(8)))  float  v8f;
typedef __attribute__((ext_vector_type(2)))  float  v2f;
typedef __attribute__((ext_vector_type(16))) __bf16 v16bf;

#define B_   4
#define C_   256
#define KK_  (C_ * 9)   // 2304 = C*3*3, K dimension of deform GEMM

// ---------------- K0: f32 -> bf16 weight cast ----------------
__global__ void cvt_bf16_kernel(const float* __restrict__ src, __bf16* __restrict__ dst, int n) {
  int i = blockIdx.x * blockDim.x + threadIdx.x;
  if (i < n) dst[i] = (__bf16)src[i];
}

// ---------------- K1: 3x3 offset conv, 256 -> 18 channels ----------------
__global__ void off_conv_kernel(const float* __restrict__ x, const float* __restrict__ w_off,
                                const float* __restrict__ b_off, float* __restrict__ off,
                                int H, int W) {
  int HW = H * W;
  int idx = blockIdx.x * blockDim.x + threadIdx.x;
  if (idx >= B_ * HW) return;
  int b = idx / HW, p = idx - b * HW, h = p / W, w = p - h * W;
  float acc[18];
  #pragma unroll
  for (int o = 0; o < 18; ++o) acc[o] = b_off[o];
  for (int c = 0; c < C_; ++c) {
    const float* xc = x + (size_t)(b * C_ + c) * HW;
    float xv[9];
    #pragma unroll
    for (int t = 0; t < 9; ++t) {
      int yy = h - 1 + t / 3, xx = w - 1 + t % 3;
      xv[t] = (yy >= 0 && yy < H && xx >= 0 && xx < W) ? xc[yy * W + xx] : 0.f;
    }
    #pragma unroll
    for (int o = 0; o < 18; ++o) {
      const float* wp = w_off + (size_t)(o * C_ + c) * 9;
      #pragma unroll
      for (int t = 0; t < 9; ++t) acc[o] += xv[t] * wp[t];
    }
  }
  for (int o = 0; o < 18; ++o) off[(size_t)(b * 18 + o) * HW + p] = acc[o];
}

// ---------------- K2: fused bilinear sampling + deform GEMM (bf16 WMMA) ----------------
// d[o, m] = b_def[o] + sum_{kk} sampled[m, kk] * w_def[o, kk],  kk = c*9+tap, m = pixel
// Double-buffered LDS A-tile: one barrier per K-step, sampling overlaps WMMA issue.
__global__ __launch_bounds__(256) void deform_gemm_kernel(
    const float* __restrict__ x, const float* __restrict__ off,
    const __bf16* __restrict__ wdef, const float* __restrict__ bdef,
    __bf16* __restrict__ dmat, int H, int W) {
  const int HW = H * W, Mtot = B_ * HW;
  __shared__ __bf16 As[2][16 * 32];
  __shared__ float spy[144], spx[144];
  int tid = threadIdx.x, lane = tid & 31, wid = tid >> 5;
  int mbase = blockIdx.x * 16;
  // precompute sample coordinates for the 16 pixels x 9 taps of this tile
  if (tid < 144) {
    int m = tid / 9, tap = tid - m * 9;
    int mg = mbase + m;
    int b = mg / HW, p = mg - b * HW, h = p / W, w = p - h * W;
    float dy = off[(size_t)(b * 18 + tap * 2    ) * HW + p];
    float dx = off[(size_t)(b * 18 + tap * 2 + 1) * HW + p];
    spy[tid] = (float)(h - 1 + tap / 3) + dy;
    spx[tid] = (float)(w - 1 + tap % 3) + dx;
  }

  // cooperative bilinear sampling of one 16x32 A tile into LDS buffer `buf`
  auto fill_tile = [&](int buf, int kc) {
    for (int e = tid; e < 512; e += 256) {
      int m = e >> 5, t = e & 31;
      int kk = kc * 32 + t;
      int c = kk / 9, tap = kk - c * 9;
      int mg = mbase + m;
      int b = mg / HW;
      float py = spy[m * 9 + tap], px = spx[m * 9 + tap];
      float y0f = floorf(py), x0f = floorf(px);
      float fy = py - y0f, fx = px - x0f;
      int y0 = (int)y0f, x0 = (int)x0f, y1 = y0 + 1, x1 = x0 + 1;
      const float* xc = x + (size_t)(b * C_ + c) * HW;
      bool yv0 = (unsigned)y0 < (unsigned)H, yv1 = (unsigned)y1 < (unsigned)H;
      bool xv0 = (unsigned)x0 < (unsigned)W, xv1 = (unsigned)x1 < (unsigned)W;
      float val = 0.f;
      if (yv0 && xv0) val += (1.f - fy) * (1.f - fx) * xc[y0 * W + x0];
      if (yv0 && xv1) val += (1.f - fy) * fx         * xc[y0 * W + x1];
      if (yv1 && xv0) val += fy * (1.f - fx)         * xc[y1 * W + x0];
      if (yv1 && xv1) val += fy * fx                 * xc[y1 * W + x1];
      As[buf][e] = (__bf16)val;
    }
  };

  __syncthreads();          // spy/spx visible
  fill_tile(0, 0);
  __syncthreads();          // first tile ready

  v8f acc0 = {}, acc1 = {};
  int ocol = wid * 32 + (lane & 15);   // wave handles output-channel tiles [wid*32,+16) and [+16,+32)
  const int NKC = KK_ / 32;            // 72
  for (int kc = 0; kc < NKC; ++kc) {
    int cur = kc & 1;
    // A fragment: M = lane&15; K contiguous per lane half (matches B)
    v16bf a, bb0, bb1;
    int mr = lane & 15;
    #pragma unroll
    for (int j = 0; j < 16; ++j)
      a[j] = As[cur][mr * 32 + j + ((lane >> 4) << 4)];
    // prefetch the next A tile into the other buffer (overlaps with WMMA)
    if (kc + 1 < NKC) fill_tile(cur ^ 1, kc + 1);
    // B fragment: w_def rows (N = output channel), K contiguous per lane half
    const __bf16* w0 = wdef + (size_t)ocol * KK_ + kc * 32 + ((lane >> 4) << 4);
    const __bf16* w1 = w0 + (size_t)16 * KK_;
    #pragma unroll
    for (int j = 0; j < 16; ++j) { bb0[j] = w0[j]; bb1[j] = w1[j]; }
    acc0 = __builtin_amdgcn_wmma_f32_16x16x32_bf16(false, a, false, bb0, (short)0, acc0, false, false);
    acc1 = __builtin_amdgcn_wmma_f32_16x16x32_bf16(false, a, false, bb1, (short)0, acc1, false, false);
    __syncthreads();
  }
  // epilogue: add bias, store D as bf16 [C, Mtot] for the QKV GEMM
  float bz0 = bdef[ocol], bz1 = bdef[ocol + 16];
  #pragma unroll
  for (int vv = 0; vv < 8; ++vv) {
    int mrow = mbase + vv + ((lane >> 4) << 3);
    dmat[(size_t)ocol * Mtot + mrow]        = (__bf16)(acc0[vv] + bz0);
    dmat[(size_t)(ocol + 16) * Mtot + mrow] = (__bf16)(acc1[vv] + bz1);
  }
}

// ---------------- K3: Q/K/V 1x1 projection GEMM (bf16 WMMA) ----------------
// out[o, m] = bias[o] + sum_c W[o,c] * D[c,m];  blockIdx.y selects q/k/v.
// The 32x16 D tile is shared by all 8 waves -> stage it once per block in LDS (double-buffered).
__global__ __launch_bounds__(256) void qkv_gemm_kernel(
    const __bf16* __restrict__ dmat,
    const __bf16* __restrict__ wq, const __bf16* __restrict__ wk, const __bf16* __restrict__ wv,
    const float* __restrict__ bq, const float* __restrict__ bk, const float* __restrict__ bv,
    float* __restrict__ qo, float* __restrict__ ko, float* __restrict__ vo, int Mtot) {
  __shared__ __bf16 Ds[2][32 * 16];
  int tid = threadIdx.x, lane = tid & 31, wid = tid >> 5;
  const __bf16* Wm; const float* bias; float* outp;
  if (blockIdx.y == 0)      { Wm = wq; bias = bq; outp = qo; }
  else if (blockIdx.y == 1) { Wm = wk; bias = bk; outp = ko; }
  else                      { Wm = wv; bias = bv; outp = vo; }
  int pbase = blockIdx.x * 16;
  int orow = wid * 32 + (lane & 15);

  auto fill_ds = [&](int buf, int kc) {
    for (int e = tid; e < 512; e += 256) {
      int kk = e >> 4, p = e & 15;
      Ds[buf][e] = dmat[(size_t)(kc * 32 + kk) * Mtot + pbase + p];
    }
  };
  fill_ds(0, 0);
  __syncthreads();

  v8f acc0 = {}, acc1 = {};
  const int NKC = C_ / 32;   // 8
  for (int kc = 0; kc < NKC; ++kc) {
    int cur = kc & 1;
    v16bf a0, a1, bb;
    // B fragment from LDS (lanes 0-15: K=0..15, lanes 16-31: K=16..31)
    #pragma unroll
    for (int j = 0; j < 16; ++j) {
      int kidx = j + ((lane >> 4) << 4);
      bb[j] = Ds[cur][kidx * 16 + (lane & 15)];
    }
    if (kc + 1 < NKC) fill_ds(cur ^ 1, kc + 1);
    // A fragments: weight rows
    const __bf16* wr0 = Wm + (size_t)orow * C_ + kc * 32;
    const __bf16* wr1 = wr0 + (size_t)16 * C_;
    #pragma unroll
    for (int j = 0; j < 16; ++j) {
      int kidx = j + ((lane >> 4) << 4);
      a0[j] = wr0[kidx]; a1[j] = wr1[kidx];
    }
    acc0 = __builtin_amdgcn_wmma_f32_16x16x32_bf16(false, a0, false, bb, (short)0, acc0, false, false);
    acc1 = __builtin_amdgcn_wmma_f32_16x16x32_bf16(false, a1, false, bb, (short)0, acc1, false, false);
    __syncthreads();
  }
  #pragma unroll
  for (int vv = 0; vv < 8; ++vv) {
    int o0 = wid * 32 + vv + ((lane >> 4) << 3);
    int o1 = o0 + 16;
    int m  = pbase + (lane & 15);
    outp[(size_t)o0 * Mtot + m] = acc0[vv] + bias[o0];
    outp[(size_t)o1 * Mtot + m] = acc1[vv] + bias[o1];
  }
}

// ---------------- K4: per-(b,c) row attention with f32 WMMA (16x16x4) ----------------
// K/V are staged into LDS with CDNA5 async memory->LDS loads (ASYNCcnt tracked).
template <int HH>
__global__ void attn_kernel(const float* __restrict__ q, const float* __restrict__ k,
                            const float* __restrict__ v, float* __restrict__ out) {
  constexpr int WW = HH, HW = HH * WW;
  __shared__ float Ksh[HW];
  __shared__ float Vsh[HW];
  __shared__ float Ssh[HH * HH];
  int bc = blockIdx.x;
  int b = bc / C_, c = bc - b * C_;
  size_t base = (size_t)(c * B_ + b) * HW;
  int tid = threadIdx.x, lane = tid & 31, wvi = tid >> 5;

  // async global->LDS staging of K and V (b128 chunks, one per 4 floats)
  {
    unsigned kl = (unsigned)(uintptr_t)&Ksh[0];   // low 32 bits of generic ptr = LDS byte address
    unsigned vl = (unsigned)(uintptr_t)&Vsh[0];
    const float* kg = k + base;
    const float* vg = v + base;
    for (int i = tid * 4; i < HW; i += blockDim.x * 4) {
      unsigned lo = (unsigned)i * 4u;
      asm volatile("global_load_async_to_lds_b128 %0, %1, off"
                   :: "v"(kl + lo), "v"(kg + i) : "memory");
      asm volatile("global_load_async_to_lds_b128 %0, %1, off"
                   :: "v"(vl + lo), "v"(vg + i) : "memory");
    }
    asm volatile("s_wait_asynccnt 0x0" ::: "memory");
  }
  __syncthreads();

  int hbase = wvi * 16;
  int hrow  = hbase + (lane & 15);
  const float scl = 0.0625f;   // DIM^-0.5 = 1/16
  // S = Q * K^T  (f32 WMMA, K-chunks of 4 over W)
  constexpr int NJT = HH / 16;
  v8f sacc[NJT] = {};
  for (int ww = 0; ww < WW; ww += 4) {
    int kq = ww + ((lane >> 4) << 1);          // lanes 16-31 take K = +2, +3
    v2f a; a.x = q[base + (size_t)hrow * WW + kq];
           a.y = q[base + (size_t)hrow * WW + kq + 1];
    #pragma unroll
    for (int jt = 0; jt < NJT; ++jt) {
      int j = jt * 16 + (lane & 15);
      v2f bb; bb.x = Ksh[j * WW + kq]; bb.y = Ksh[j * WW + kq + 1];
      sacc[jt] = __builtin_amdgcn_wmma_f32_16x16x4_f32(false, a, false, bb, (short)0, sacc[jt], false, false);
    }
  }
  #pragma unroll
  for (int jt = 0; jt < NJT; ++jt)
    #pragma unroll
    for (int vvi = 0; vvi < 8; ++vvi) {
      int r = hbase + vvi + ((lane >> 4) << 3);
      Ssh[r * HH + jt * 16 + (lane & 15)] = sacc[jt][vvi] * scl;
    }
  __syncthreads();
  // softmax over j, one row per thread (blockDim = 2*HH >= HH)
  if (tid < HH) {
    float* row = Ssh + tid * HH;
    float mx = -3.402823466e38f;
    for (int j = 0; j < HH; ++j) mx = fmaxf(mx, row[j]);
    float s = 0.f;
    for (int j = 0; j < HH; ++j) { float e = __expf(row[j] - mx); row[j] = e; s += e; }
    float inv = 1.f / s;
    for (int j = 0; j < HH; ++j) row[j] *= inv;
  }
  __syncthreads();
  // O = P * V
  constexpr int NWT = WW / 16;
  v8f oacc[NWT] = {};
  for (int jj = 0; jj < HH; jj += 4) {
    int kq = jj + ((lane >> 4) << 1);
    v2f a; a.x = Ssh[hrow * HH + kq]; a.y = Ssh[hrow * HH + kq + 1];
    #pragma unroll
    for (int wt = 0; wt < NWT; ++wt) {
      int wcol = wt * 16 + (lane & 15);
      v2f bb; bb.x = Vsh[kq * WW + wcol]; bb.y = Vsh[(kq + 1) * WW + wcol];
      oacc[wt] = __builtin_amdgcn_wmma_f32_16x16x4_f32(false, a, false, bb, (short)0, oacc[wt], false, false);
    }
  }
  size_t obase = (size_t)(b * C_ + c) * HW;   // output layout [B, C, H, W]
  #pragma unroll
  for (int wt = 0; wt < NWT; ++wt)
    #pragma unroll
    for (int vvi = 0; vvi < 8; ++vvi) {
      int r = hbase + vvi + ((lane >> 4) << 3);
      out[obase + (size_t)r * WW + wt * 16 + (lane & 15)] = oacc[wt][vvi];
    }
}

// ---------------- launcher ----------------
extern "C" void kernel_launch(void* const* d_in, const int* in_sizes, int n_in,
                              void* d_out, int out_size, void* d_ws, size_t ws_size,
                              hipStream_t stream) {
  const float* s[3]   = {(const float*)d_in[0], (const float*)d_in[1], (const float*)d_in[2]};
  const float* w_off  = (const float*)d_in[3];
  const float* b_off  = (const float*)d_in[4];
  const float* w_def  = (const float*)d_in[5];
  const float* b_def  = (const float*)d_in[6];
  const float* wq     = (const float*)d_in[7];
  const float* bq     = (const float*)d_in[8];
  const float* wk     = (const float*)d_in[9];
  const float* bk     = (const float*)d_in[10];
  const float* wv     = (const float*)d_in[11];
  const float* bv     = (const float*)d_in[12];
  float* out = (float*)d_out;

  char* ws = (char*)d_ws;
  size_t cur = 0;
  auto take = [&](size_t bytes) { char* p = ws + cur; cur = (cur + bytes + 255) & ~(size_t)255; return p; };
  float*  off_ws  = (float*)take((size_t)B_ * 18 * 64 * 64 * 4);   // 1.13 MB
  __bf16* wdef_bf = (__bf16*)take((size_t)C_ * KK_ * 2);           // 1.13 MB
  __bf16* wq_bf   = (__bf16*)take((size_t)C_ * C_ * 2);
  __bf16* wk_bf   = (__bf16*)take((size_t)C_ * C_ * 2);
  __bf16* wv_bf   = (__bf16*)take((size_t)C_ * C_ * 2);
  __bf16* d_bf    = (__bf16*)take((size_t)C_ * B_ * 64 * 64 * 2);  // 8 MB
  float*  q_ws    = (float*)take((size_t)C_ * B_ * 64 * 64 * 4);   // 16 MB each
  float*  k_ws    = (float*)take((size_t)C_ * B_ * 64 * 64 * 4);
  float*  v_ws    = (float*)take((size_t)C_ * B_ * 64 * 64 * 4);   // total ~62 MB

  // one-time weight casts to bf16
  cvt_bf16_kernel<<<(C_ * KK_ + 255) / 256, 256, 0, stream>>>(w_def, wdef_bf, C_ * KK_);
  cvt_bf16_kernel<<<(C_ * C_ + 255) / 256, 256, 0, stream>>>(wq, wq_bf, C_ * C_);
  cvt_bf16_kernel<<<(C_ * C_ + 255) / 256, 256, 0, stream>>>(wk, wk_bf, C_ * C_);
  cvt_bf16_kernel<<<(C_ * C_ + 255) / 256, 256, 0, stream>>>(wv, wv_bf, C_ * C_);

  const int Hs[3] = {64, 32, 16};
  const size_t outbase[3] = {0, (size_t)B_ * C_ * 64 * 64,
                             (size_t)B_ * C_ * 64 * 64 + (size_t)B_ * C_ * 32 * 32};
  for (int i = 0; i < 3; ++i) {
    int H = Hs[i], W = H, HW = H * W, Mtot = B_ * HW;
    off_conv_kernel<<<(B_ * HW + 255) / 256, 256, 0, stream>>>(s[i], w_off, b_off, off_ws, H, W);
    deform_gemm_kernel<<<Mtot / 16, 256, 0, stream>>>(s[i], off_ws, wdef_bf, b_def, d_bf, H, W);
    qkv_gemm_kernel<<<dim3(Mtot / 16, 3), 256, 0, stream>>>(d_bf, wq_bf, wk_bf, wv_bf,
                                                            bq, bk, bv, q_ws, k_ws, v_ws, Mtot);
    float* o = out + outbase[i];
    if (H == 64)      attn_kernel<64><<<B_ * C_, 128, 0, stream>>>(q_ws, k_ws, v_ws, o);
    else if (H == 32) attn_kernel<32><<<B_ * C_,  64, 0, stream>>>(q_ws, k_ws, v_ws, o);
    else              attn_kernel<16><<<B_ * C_,  32, 0, stream>>>(q_ws, k_ws, v_ws, o);
  }
  (void)in_sizes; (void)n_in; (void)out_size; (void)ws_size;
}
